// GaussianSelfAttention_1975684956773
// MI455X (gfx1250) — matrix operand order repaired and checked by reading, not compiled, not run on hardware
//
#include <hip/hip_runtime.h>
#include <hip/hip_bf16.h>

// ---------------------------------------------------------------------------
// Problem constants
// ---------------------------------------------------------------------------
#define DIMX     768
#define N_IMGS   1024
#define N_PATCH  196
#define GRIDW    14
#define BATCH    64
#define ROWS     (BATCH * N_PATCH)        // 12544 rows of X / Q / K / V
#define NELEM_X  ((size_t)ROWS * DIMX)    // 9,633,792
#define NELEM_W  ((size_t)DIMX * DIMX)    // 589,824

// Workspace layout (bytes)
#define XBF_OFF   ((size_t)0)                         // bf16 X          : ROWS x DIM
#define WBF_OFF   (XBF_OFF + NELEM_X * 2)             // bf16 Wq|Wk|Wv   : 3 x DIM x DIM
#define QKV_OFF   (WBF_OFF + 3 * NELEM_W * 2)         // f32  Q|K|V      : 3 x ROWS x DIM
#define IDX_OFF   (QKV_OFF + 3 * NELEM_X * 4)         // int4 indices    : ROWS
// total ~138.6 MB

typedef __attribute__((ext_vector_type(16))) __bf16 v16bf;
typedef __attribute__((ext_vector_type(8)))  __bf16 v8bf;
typedef __attribute__((ext_vector_type(8)))  float  v8f;

// WMMA operand fragment: 8 VGPRs; halves fillable by two b128 loads with no
// repacking, or the whole thing by one (split) 32B load.
union Frag {
    v16bf v;
    struct { v8bf lo; v8bf hi; } h;
};

// ---------------------------------------------------------------------------
// fp32 -> bf16 (round-to-nearest-even) elementwise convert
// ---------------------------------------------------------------------------
__global__ void cvt_bf16_kernel(const float* __restrict__ src,
                                unsigned short* __restrict__ dst, int n) {
    int i = blockIdx.x * blockDim.x + threadIdx.x;
    if (i < n) {
        unsigned u = __float_as_uint(src[i]);
        u += 0x7FFFu + ((u >> 16) & 1u);
        dst[i] = (unsigned short)(u >> 16);
    }
}

// ---------------------------------------------------------------------------
// QKV GEMM: C[z] = X (ROWSxDIM, bf16) * W[z]^T (DIM x DIM, bf16, stored (e,d))
//           + bias[z], f32 accumulate via v_wmma_f32_16x16x32_bf16.
//
// 4x4 register-blocked: one wave -> one 64x64 C tile.
// Per 32-wide K step: 4 A-frags (2 x b128 each) + 4 B-frags (2 x b128 each)
// feed 16 WMMAs => 1 b128 load per WMMA. K-loop = 768/32 = 24 steps,
// 384 WMMAs per wave, 16 v8f accumulators (~210 VGPRs/wave).
//
// grid = (768/64 = 12, ROWS/64 = 196, 3), block = 32 (single wave, no barrier)
// ---------------------------------------------------------------------------
__global__ __launch_bounds__(32)
void qkv_wmma_kernel(const __bf16* __restrict__ X,
                     const __bf16* __restrict__ Wb,   // 3 matrices contiguous
                     const float*  __restrict__ bq,
                     const float*  __restrict__ bk,
                     const float*  __restrict__ bv,
                     float* __restrict__ QKV) {       // 3 outputs contiguous
    const int lane = threadIdx.x & 31;
    const int g    = lane >> 4;                       // lane half
    const int l16  = lane & 15;
    const int tn   = blockIdx.x;                      // N supertile: 0..11
    const int tm   = blockIdx.y;                      // M supertile: 0..195
    const int z    = blockIdx.z;                      // 0=q 1=k 2=v

    const __bf16* W    = Wb + (size_t)z * NELEM_W;
    const float*  bias = (z == 0) ? bq : (z == 1) ? bk : bv;

    // A: rows of X; B: columns of W^T == rows of W (contiguous in d = K)
    const __bf16* arow[4];
    const __bf16* brow[4];
    #pragma unroll
    for (int i = 0; i < 4; ++i) {
        arow[i] = X + (size_t)(tm * 64 + i * 16 + l16) * DIMX;
        brow[i] = W + (size_t)(tn * 64 + i * 16 + l16) * DIMX;
        // gfx1250 global_prefetch_b8: warm the streamed rows
        __builtin_prefetch(arow[i], 0, 1);
        __builtin_prefetch(brow[i], 0, 1);
    }

    v8f acc[4][4] = {};

    for (int k0 = 0; k0 < DIMX; k0 += 32) {
        Frag A[4], B[4];
        #pragma unroll
        for (int i = 0; i < 4; ++i) {
            // A 16-bit layout: lane (m,g) holds K = g*8+{0..7} then 16+g*8+{0..7}
            A[i].h.lo = *(const v8bf*)(arow[i] + k0 + g * 8);
            A[i].h.hi = *(const v8bf*)(arow[i] + k0 + 16 + g * 8);
            // B 16-bit layout: lane (n,g) holds K = g*8+{0..7} then 16+g*8+{0..7}
            B[i].h.lo = *(const v8bf*)(brow[i] + k0 + g * 8);
            B[i].h.hi = *(const v8bf*)(brow[i] + k0 + 16 + g * 8);
        }
        #pragma unroll
        for (int i = 0; i < 4; ++i)
            #pragma unroll
            for (int j = 0; j < 4; ++j)
                // (neg_a, A, neg_b, B, c_mod, C, reuse_a, reuse_b)
                acc[i][j] = __builtin_amdgcn_wmma_f32_16x16x32_bf16(
                    false, A[i].v, false, B[j].v, (short)0, acc[i][j], false, false);
    }

    // C/D layout: VGPR r -> M = g*8 + r, N = l16
    #pragma unroll
    for (int j = 0; j < 4; ++j) {
        const float bval = bias[tn * 64 + j * 16 + l16];
        float* ocol = QKV + (size_t)z * NELEM_X + (size_t)(tn * 64 + j * 16 + l16);
        #pragma unroll
        for (int i = 0; i < 4; ++i) {
            #pragma unroll
            for (int r = 0; r < 8; ++r) {
                int m = tm * 64 + i * 16 + g * 8 + r;
                ocol[(size_t)m * DIMX] = acc[i][j][r] + bval;
            }
        }
    }
}

// ---------------------------------------------------------------------------
// Per (b,s): Gaussian-sample coords, 4 bilinear-neighbor indices (clipped)
// ---------------------------------------------------------------------------
__global__ void coord_kernel(const int* __restrict__ img_ids,
                             const float* __restrict__ eps,       // (B,2,196)
                             const float* __restrict__ avgs,      // (N_IMGS,2,196)
                             const float* __restrict__ stds,      // (N_IMGS,2,196)
                             int4* __restrict__ idx) {
    int t = blockIdx.x * blockDim.x + threadIdx.x;
    if (t >= ROWS) return;
    int b = t / N_PATCH, s = t - b * N_PATCH;
    int id = img_ids[b];
    size_t ia = (size_t)id * 2 * N_PATCH + s;
    size_t ie = (size_t)b  * 2 * N_PATCH + s;
    float kx = avgs[ia]           + stds[ia]           * eps[ie];
    float ky = avgs[ia + N_PATCH] + stds[ia + N_PATCH] * eps[ie + N_PATCH];
    float kx1 = ceilf(kx), kx2 = floorf(kx);
    float ky1 = ceilf(ky), ky2 = floorf(ky);
    // clip in float to [0, N_PATCH-1], then truncate (values >= 0)
    float f0 = fminf(fmaxf((float)GRIDW * ky1 + kx1, 0.f), (float)(N_PATCH - 1));
    float f1 = fminf(fmaxf((float)GRIDW * ky1 + kx2, 0.f), (float)(N_PATCH - 1));
    float f2 = fminf(fmaxf((float)GRIDW * ky2 + kx1, 0.f), (float)(N_PATCH - 1));
    float f3 = fminf(fmaxf((float)GRIDW * ky2 + kx2, 0.f), (float)(N_PATCH - 1));
    idx[t] = make_int4((int)f0, (int)f1, (int)f2, (int)f3);
}

// ---------------------------------------------------------------------------
// Per (b,s,e): a_j = q_e * k[idx_j]_e ; softmax over j=0..3 ; out = sum w_j v_j
// grid = (DIM/256, ROWS), block = 256; e = coalesced fastest dim
// ---------------------------------------------------------------------------
__global__ __launch_bounds__(256)
void attn_kernel(const float* __restrict__ QKV,
                 const int4* __restrict__ idx,
                 float* __restrict__ out) {
    int e  = blockIdx.x * 256 + threadIdx.x;   // 0..767
    int bs = blockIdx.y;                       // 0..12543
    int b  = bs / N_PATCH;

    const float* Q = QKV;
    const float* K = QKV + NELEM_X;
    const float* V = QKV + 2 * NELEM_X;

    int4 id = idx[bs];
    float q = Q[(size_t)bs * DIMX + e];
    size_t base = (size_t)b * N_PATCH * DIMX + e;

    float k0 = K[base + (size_t)id.x * DIMX];
    float k1 = K[base + (size_t)id.y * DIMX];
    float k2 = K[base + (size_t)id.z * DIMX];
    float k3 = K[base + (size_t)id.w * DIMX];
    float v0 = V[base + (size_t)id.x * DIMX];
    float v1 = V[base + (size_t)id.y * DIMX];
    float v2 = V[base + (size_t)id.z * DIMX];
    float v3 = V[base + (size_t)id.w * DIMX];

    float a0 = q * k0, a1 = q * k1, a2 = q * k2, a3 = q * k3;
    float m  = fmaxf(fmaxf(a0, a1), fmaxf(a2, a3));
    float e0 = __expf(a0 - m), e1 = __expf(a1 - m);
    float e2 = __expf(a2 - m), e3 = __expf(a3 - m);
    float inv = 1.0f / (e0 + e1 + e2 + e3);
    out[(size_t)bs * DIMX + e] = (e0 * v0 + e1 * v1 + e2 * v2 + e3 * v3) * inv;
}

// ---------------------------------------------------------------------------
// Launcher
// ---------------------------------------------------------------------------
extern "C" void kernel_launch(void* const* d_in, const int* in_sizes, int n_in,
                              void* d_out, int out_size, void* d_ws, size_t ws_size,
                              hipStream_t stream) {
    const float* x       = (const float*)d_in[0];
    // d_in[1] = mask (unused by the reference math)
    const int*   img_ids = (const int*)  d_in[2];
    const float* eps     = (const float*)d_in[3];
    const float* Wq      = (const float*)d_in[4];
    const float* bq      = (const float*)d_in[5];
    const float* Wk      = (const float*)d_in[6];
    const float* bk      = (const float*)d_in[7];
    const float* Wv      = (const float*)d_in[8];
    const float* bv      = (const float*)d_in[9];
    const float* avgs    = (const float*)d_in[10];
    const float* stds    = (const float*)d_in[11];

    char* ws = (char*)d_ws;
    unsigned short* xbf = (unsigned short*)(ws + XBF_OFF);
    unsigned short* wbf = (unsigned short*)(ws + WBF_OFF);
    float*          qkv = (float*)         (ws + QKV_OFF);
    int4*           idx = (int4*)          (ws + IDX_OFF);

    // 1) bf16 conversions
    {
        int n = (int)NELEM_X;
        cvt_bf16_kernel<<<(n + 255) / 256, 256, 0, stream>>>(x, xbf, n);
        int nw = (int)NELEM_W;
        cvt_bf16_kernel<<<(nw + 255) / 256, 256, 0, stream>>>(Wq, wbf + 0 * NELEM_W, nw);
        cvt_bf16_kernel<<<(nw + 255) / 256, 256, 0, stream>>>(Wk, wbf + 1 * NELEM_W, nw);
        cvt_bf16_kernel<<<(nw + 255) / 256, 256, 0, stream>>>(Wv, wbf + 2 * NELEM_W, nw);
    }

    // 2) bilinear indices
    coord_kernel<<<(ROWS + 255) / 256, 256, 0, stream>>>(img_ids, eps, avgs, stds, idx);

    // 3) QKV GEMM via WMMA, 64x64 C tile per wave
    {
        dim3 grid(DIMX / 64, ROWS / 64, 3);
        qkv_wmma_kernel<<<grid, 32, 0, stream>>>((const __bf16*)xbf, (const __bf16*)wbf,
                                                 bq, bk, bv, qkv);
    }

    // 4) gather + per-dim softmax-of-4 + weighted sum
    {
        dim3 grid(DIMX / 256, ROWS);
        attn_kernel<<<grid, 256, 0, stream>>>(qkv, idx, (float*)d_out);
    }
}
